// FcModel_23330262352100
// MI455X (gfx1250) — matrix operand order repaired and checked by reading, not compiled, not run on hardware
//
#include <hip/hip_runtime.h>

// ---------------------------------------------------------------------------
// GraphConv x3 (+ self loops, symmetric norm) + MLP head for MI455X (gfx1250).
// Edge aggregation via f32 global atomics (L2-resident), dense node GEMMs via
// v_wmma_f32_16x16x32_f16 (f32 accumulate), wave32 layouts per CDNA5 ISA 7.12.2.
// A fragments: two aligned b128 loads; B pre-swizzled fragment-major (b128 x2).
// ---------------------------------------------------------------------------

typedef __attribute__((ext_vector_type(16))) _Float16 v16h;
typedef __attribute__((ext_vector_type(8)))  _Float16 v8h;
typedef __attribute__((ext_vector_type(8)))  float    v8f;

#define FIN  6
#define HID  128
#define GOUT 64

// ---------------- small utility kernels ----------------

__global__ void k_init_deg(float* dout, float* din, int n) {
  int i = blockIdx.x * blockDim.x + threadIdx.x;
  if (i < n) { dout[i] = 1.0f; din[i] = 1.0f; }   // self loop counts as 1
}

__global__ void k_zero(float* p, int n) {
  int i = blockIdx.x * blockDim.x + threadIdx.x;
  if (i < n) p[i] = 0.0f;
}

__global__ void k_deg_count(const int* __restrict__ src, const int* __restrict__ dst,
                            float* dout, float* din, int nE) {
  int e = blockIdx.x * blockDim.x + threadIdx.x;
  if (e < nE) {
    atomicAdd(&dout[src[e]], 1.0f);
    atomicAdd(&din[dst[e]], 1.0f);
  }
}

__global__ void k_cnorm(float* c, int n) {   // in place: deg -> clip(deg,1)^-0.5
  int i = blockIdx.x * blockDim.x + threadIdx.x;
  if (i < n) c[i] = rsqrtf(fmaxf(c[i], 1.0f));
}

// Swizzle f32 weights [K x N] row-major into fragment-major fp16:
//   Bs[ ((kc*nTN + nt)*32 + lane)*16 + i ]  where fragment element i of lane
//   holds B[kc*32 + (i&7) + 16*(i>=8) + 8*(lane>=16)][nt*16 + (lane&15)]   (ISA 7.12.2 B layout)
__global__ void k_swizzleB(const float* __restrict__ W, _Float16* __restrict__ Bs,
                           int K, int N) {
  int idx = blockIdx.x * blockDim.x + threadIdx.x;
  if (idx >= K * N) return;
  int i    = idx & 15;
  int lane = (idx >> 4) & 31;
  int t    = idx >> 9;
  int nTN  = N >> 4;
  int nt   = t % nTN;
  int kc   = t / nTN;
  int Kpos = kc * 32 + (i & 7) + ((i >> 3) << 4) + ((lane >> 4) << 3);
  int Npos = nt * 16 + (lane & 15);
  Bs[idx] = (_Float16)W[Kpos * N + Npos];
}

// ---------------- layer 1: scale+init, scatter(f=6), project ----------------

__global__ void k_hs1(const float* __restrict__ nf, const float* __restrict__ c_src,
                      float* hs1, float* agg1, int n6) {
  int i = blockIdx.x * blockDim.x + threadIdx.x;
  if (i < n6) {
    float v = nf[i] * c_src[i / FIN];
    hs1[i] = v;          // immutable gather source
    agg1[i] = v;         // accumulator initialized with self-loop message
  }
}

__global__ void k_scatter6(const int* __restrict__ src, const int* __restrict__ dst,
                           const float* __restrict__ hs, float* agg, int nE) {
  int e = blockIdx.x * blockDim.x + threadIdx.x;
  if (e >= nE) return;
  int s = src[e] * FIN, d = dst[e] * FIN;
#pragma unroll
  for (int f = 0; f < FIN; ++f) atomicAdd(&agg[d + f], hs[s + f]);
}

// h1 = relu(agg1*c_dst @ Wg1 + bg1); emit hs2 = h1*c_src and agg2 init = hs2
__global__ void k_l1_post(const float* __restrict__ agg1, const float* __restrict__ c_src,
                          const float* __restrict__ c_dst, const float* __restrict__ Wg1,
                          const float* __restrict__ bg1, float* hs2, float* agg2, int n) {
  int node = blockIdx.x;
  int j = threadIdx.x;                 // 0..127
  if (node >= n) return;
  float cd = c_dst[node];
  float s = bg1[j];
#pragma unroll
  for (int k = 0; k < FIN; ++k)
    s += agg1[node * FIN + k] * cd * Wg1[k * HID + j];
  s = fmaxf(s, 0.0f) * c_src[node];
  hs2[node * HID + j] = s;
  agg2[node * HID + j] = s;
}

// ---------------- wide scatters (float4 per lane, edge uniform per 32 / 16 lanes) ----------------

__global__ void k_scatter128(const int* __restrict__ src, const int* __restrict__ dst,
                             const float* __restrict__ hs, float* agg, int nE) {
  unsigned tid = blockIdx.x * blockDim.x + threadIdx.x;
  int e = (int)(tid >> 5);             // 32 lanes cover 128 floats
  if (e >= nE) return;
  int c = ((int)tid & 31) * 4;
  int s = src[e], d = dst[e];
  const float4 v = *(const float4*)(hs + (size_t)s * HID + c);
  float* p = agg + (size_t)d * HID + c;
  atomicAdd(p + 0, v.x); atomicAdd(p + 1, v.y);
  atomicAdd(p + 2, v.z); atomicAdd(p + 3, v.w);
}

__global__ void k_scatter64(const int* __restrict__ src, const int* __restrict__ dst,
                            const float* __restrict__ hs, float* agg, int nE) {
  unsigned tid = blockIdx.x * blockDim.x + threadIdx.x;
  int e = (int)(tid >> 4);             // 16 lanes cover 64 floats
  if (e >= nE) return;
  int c = ((int)tid & 15) * 4;
  int s = src[e], d = dst[e];
  const float4 v = *(const float4*)(hs + (size_t)s * GOUT + c);
  float* p = agg + (size_t)d * GOUT + c;
  atomicAdd(p + 0, v.x); atomicAdd(p + 1, v.y);
  atomicAdd(p + 2, v.z); atomicAdd(p + 3, v.w);
}

// agg2 * c_dst -> fp16 GEMM operand (row-major)
__global__ void k_a2h(const float* __restrict__ agg, const float* __restrict__ c_dst,
                      _Float16* __restrict__ out, int total) {
  int i = blockIdx.x * blockDim.x + threadIdx.x;
  if (i < total) out[i] = (_Float16)(agg[i] * c_dst[i >> 7]);   // row = i/128
}

// ---------------- WMMA GEMM: C[M, NTN*16] = A[M,K] @ B, fused epilogue ----------
// A fp16 row-major (lda = K); B pre-swizzled fragment-major (k_swizzleB).
// One wave handles one 16-row M tile and ALL NTN column tiles (A reuse).
template <int NTN, int K>
__global__ void k_wmma_gemm(const _Float16* __restrict__ A, const _Float16* __restrict__ Bs,
                            const float* __restrict__ bias, const float* __restrict__ rowscale,
                            float* __restrict__ outF, float* __restrict__ outF2,
                            _Float16* __restrict__ outH, int M, int relu) {
  constexpr int N = NTN * 16;
  const int lane = threadIdx.x & 31;
  const int Mt = M >> 4;
  int mt = blockIdx.x * 8 + (threadIdx.x >> 5);     // 8 waves / block, wave-uniform
  if (mt >= Mt) return;
  const int lo = lane & 15;
  const int hi = lane >> 4;                         // lane group 0/1
  const int row = mt * 16 + lo;                     // A row held by this lane
  const int kA = hi * 8;                            // A K-offset per lane group

  v8f acc[NTN];
#pragma unroll
  for (int nt = 0; nt < NTN; ++nt) acc[nt] = (v8f){};

  union Frag { v16h v; v8h h[2]; };

#pragma unroll
  for (int k0 = 0; k0 < K; k0 += 32) {
    // A fragment: lane holds K = {kA..kA+7} U {kA+16..kA+23} -> two b128 loads
    const _Float16* Ap = A + (size_t)row * K + k0 + kA;
    Frag a;
    a.h[0] = *(const v8h*)(Ap);
    a.h[1] = *(const v8h*)(Ap + 16);
#pragma unroll
    for (int nt = 0; nt < NTN; ++nt) {
      // B fragment: 32 contiguous bytes per lane in swizzled buffer
      const v16h b = *(const v16h*)(Bs + (((size_t)(k0 >> 5) * NTN + nt) * 32 + lane) * 16);
      acc[nt] = __builtin_amdgcn_wmma_f32_16x16x32_f16(
          false, a.v, false, b, (short)0, acc[nt], false, false);
    }
  }

  const int mbase = mt * 16 + hi * 8;               // D layout: M = r + 8*(lane>=16)
#pragma unroll
  for (int nt = 0; nt < NTN; ++nt) {
    const int col = nt * 16 + lo;
#pragma unroll
    for (int r = 0; r < 8; ++r) {
      int m = mbase + r;
      float v = acc[nt][r];
      if (bias)     v += bias[col];
      if (relu)     v = fmaxf(v, 0.0f);
      if (rowscale) v *= rowscale[m];
      size_t idx = (size_t)m * N + col;
      if (outF)  outF[idx]  = v;
      if (outF2) outF2[idx] = v;
      if (outH)  outH[idx]  = (_Float16)v;
    }
  }
}

// ---------------- column-sum of agg3*c_dst (graph mean, pre-bias) ----------------

__global__ void k_colsum64(const float* __restrict__ agg3, const float* __restrict__ c_dst,
                           float* accum, int n) {
  __shared__ float sm[256];
  int col = threadIdx.x & 63;
  int grp = threadIdx.x >> 6;                       // 4 row-groups per block
  float s = 0.0f;
  for (int i = blockIdx.x * 4 + grp; i < n; i += gridDim.x * 4)
    s += agg3[(size_t)i * GOUT + col] * c_dst[i];
  sm[threadIdx.x] = s;
  __syncthreads();
  if (threadIdx.x < 64) {
    float t = sm[threadIdx.x] + sm[threadIdx.x + 64] +
              sm[threadIdx.x + 128] + sm[threadIdx.x + 192];
    atomicAdd(&accum[col], t);
  }
}

// ---------------- MLP head, single block ----------------

__global__ void k_head(const float* __restrict__ x,
                       const float* __restrict__ W1, const float* __restrict__ b1,
                       const float* __restrict__ W2, const float* __restrict__ b2,
                       const float* __restrict__ W3, const float* __restrict__ b3,
                       const float* __restrict__ W4, const float* __restrict__ b4,
                       const float* __restrict__ gs_accum, const float* __restrict__ bg3,
                       float inv_n, float* out) {
  __shared__ float z[128], z2[64], z3[32];
  int t = threadIdx.x;
  if (t < 64) {
    float s = b1[t];
    for (int k = 0; k < 1024; ++k) s += x[k] * W1[k * 64 + t];
    z[t] = fmaxf(s, 0.0f);
  } else {
    int j = t - 64;
    z[t] = gs_accum[j] * inv_n + bg3[j];            // mean_nodes + bg3
  }
  __syncthreads();
  if (t < 64) {
    float s = b2[t];
    for (int k = 0; k < 128; ++k) s += z[k] * W2[k * 64 + t];
    z2[t] = fmaxf(s, 0.0f);
  }
  __syncthreads();
  if (t < 32) {
    float s = b3[t];
    for (int k = 0; k < 64; ++k) s += z2[k] * W3[k * 32 + t];
    z3[t] = fmaxf(s, 0.0f);
  }
  __syncthreads();
  if (t < 4) {
    float s = b4[t];
    for (int k = 0; k < 32; ++k) s += z3[k] * W4[k * 4 + t];
    out[t] = s;
  }
}

// ---------------------------------------------------------------------------

#define CEIL(a, b) (((a) + (b) - 1) / (b))

extern "C" void kernel_launch(void* const* d_in, const int* in_sizes, int n_in,
                              void* d_out, int out_size, void* d_ws, size_t ws_size,
                              hipStream_t stream) {
  (void)n_in; (void)out_size; (void)ws_size;
  const float* x   = (const float*)d_in[0];
  const float* nf  = (const float*)d_in[1];
  const int*   src = (const int*)d_in[2];
  const int*   dst = (const int*)d_in[3];
  const float* Wg1 = (const float*)d_in[4];
  const float* bg1 = (const float*)d_in[5];
  const float* Wg2 = (const float*)d_in[6];
  const float* bg2 = (const float*)d_in[7];
  const float* Wg3 = (const float*)d_in[8];
  const float* bg3 = (const float*)d_in[9];
  const float* W1  = (const float*)d_in[10];
  const float* b1  = (const float*)d_in[11];
  const float* W2  = (const float*)d_in[12];
  const float* b2  = (const float*)d_in[13];
  const float* W3  = (const float*)d_in[14];
  const float* b3  = (const float*)d_in[15];
  const float* W4  = (const float*)d_in[16];
  const float* b4  = (const float*)d_in[17];
  float* out = (float*)d_out;

  const int n = in_sizes[1] / FIN;   // 100000 (multiple of 16)
  const int E = in_sizes[2];         // 1600000

  // ---- carve workspace (~211 MB total) ----
  char* p = (char*)d_ws;
  auto carve = [&](size_t bytes) {
    char* r = p; p += (bytes + 255) & ~(size_t)255; return r;
  };
  float*    c_src    = (float*)   carve((size_t)n * 4);            // deg_out -> norm
  float*    c_dst    = (float*)   carve((size_t)n * 4);            // deg_in  -> norm
  float*    hs1      = (float*)   carve((size_t)n * FIN * 4);
  float*    agg1     = (float*)   carve((size_t)n * FIN * 4);
  float*    hs2      = (float*)   carve((size_t)n * HID * 4);
  float*    agg2     = (float*)   carve((size_t)n * HID * 4);
  _Float16* a2h      = (_Float16*)carve((size_t)n * HID * 2);
  _Float16* h2h      = (_Float16*)carve((size_t)n * HID * 2);
  float*    hs3      = (float*)   carve((size_t)n * GOUT * 4);
  float*    agg3     = (float*)   carve((size_t)n * GOUT * 4);
  _Float16* w2s      = (_Float16*)carve((size_t)HID * HID * 2);    // swizzled B
  _Float16* w3s      = (_Float16*)carve((size_t)HID * GOUT * 2);   // swizzled B
  float*    gs_accum = (float*)   carve(64 * 4);

  // ---- degrees & symmetric norm ----
  k_init_deg<<<CEIL(n, 256), 256, 0, stream>>>(c_src, c_dst, n);
  k_zero<<<1, 64, 0, stream>>>(gs_accum, 64);
  k_deg_count<<<CEIL(E, 256), 256, 0, stream>>>(src, dst, c_src, c_dst, E);
  k_cnorm<<<CEIL(n, 256), 256, 0, stream>>>(c_src, n);
  k_cnorm<<<CEIL(n, 256), 256, 0, stream>>>(c_dst, n);

  // ---- weight fp16 fragment-swizzles (independent, tiny) ----
  k_swizzleB<<<CEIL(HID * HID, 256), 256, 0, stream>>>(Wg2, w2s, HID, HID);
  k_swizzleB<<<CEIL(HID * GOUT, 256), 256, 0, stream>>>(Wg3, w3s, HID, GOUT);

  // ---- layer 1: aggregate on f=6, project to 128, ReLU; emit hs2/agg2 ----
  k_hs1<<<CEIL(n * FIN, 256), 256, 0, stream>>>(nf, c_src, hs1, agg1, n * FIN);
  k_scatter6<<<CEIL(E, 256), 256, 0, stream>>>(src, dst, hs1, agg1, E);
  k_l1_post<<<n, HID, 0, stream>>>(agg1, c_src, c_dst, Wg1, bg1, hs2, agg2, n);

  // ---- layer 2: aggregate on f=128, then WMMA GEMM 128x128 (+bias, ReLU) -> fp16 ----
  k_scatter128<<<CEIL(E * 32, 256), 256, 0, stream>>>(src, dst, hs2, agg2, E);
  k_a2h<<<CEIL(n * HID, 256), 256, 0, stream>>>(agg2, c_dst, a2h, n * HID);
  k_wmma_gemm<HID / 16, HID><<<CEIL(n / 16, 8), 256, 0, stream>>>(
      a2h, w2s, bg2, nullptr, nullptr, nullptr, h2h, n, 1);

  // ---- layer 3: project first (WMMA, rowscale=c_src), then aggregate on f=64 ----
  k_wmma_gemm<GOUT / 16, HID><<<CEIL(n / 16, 8), 256, 0, stream>>>(
      h2h, w3s, nullptr, c_src, hs3, agg3, nullptr, n, 0);
  k_scatter64<<<CEIL(E * 16, 256), 256, 0, stream>>>(src, dst, hs3, agg3, E);

  // ---- graph mean (pre-bias column sums) + MLP head ----
  k_colsum64<<<256, 256, 0, stream>>>(agg3, c_dst, gs_accum, n);
  k_head<<<1, 128, 0, stream>>>(x, W1, b1, W2, b2, W3, b3, W4, b4,
                                gs_accum, bg3, 1.0f / (float)n, out);
}
